// CustomAttention_28149215658264
// MI455X (gfx1250) — compile-verified, matrix operand order checked by reading, not run on hardware
//
#include <hip/hip_runtime.h>
#include <hip/hip_bf16.h>

// ---------------------------------------------------------------------------
// Non-local attention block, CDNA5 (gfx1250, wave32, WMMA, async LDS loads).
//   B=8, C=512, H=W=64 -> N=4096, DQK=64
// Flash-attention single pass (537MB attn matrix never hits HBM), bf16 WMMA
// with f32 accum, double-buffered GLOBAL_LOAD_ASYNC_TO_LDS K/V staging.
// ---------------------------------------------------------------------------

#define B_   8
#define C_   512
#define N_   4096
#define DQK_ 64

typedef __attribute__((ext_vector_type(16))) __bf16 v16bf;
typedef __attribute__((ext_vector_type(8)))  float  v8f;

// Load a 16-bit WMMA fragment (A-style layout) from a [row][k] (k-contiguous)
// buffer. Per CDNA5 ISA 7.12.2: lanes 0-15 hold K=0..7 (VGPR0-3) and K=16..23
// (VGPR4-7); lanes 16-31 hold K=8..15 and K=24..31 -> two 16-byte loads.
__device__ __forceinline__ v16bf load_frag16(const __bf16* base, int row, int ld, int lane) {
    const int kb = (lane >> 4) << 3;   // 0 or 8
    const __bf16* p = base + row * ld + kb;
    union { v16bf v; uint4 u[2]; } f;
    f.u[0] = *(const uint4*)(p);
    f.u[1] = *(const uint4*)(p + 16);
    return f.v;
}

// Async global->LDS 16-byte copy (GVS mode: saddr base + 32-bit vgpr offset).
// Flat addresses of __shared__ objects carry the LDS byte offset in the low
// 32 bits (ISA 10.2 aperture mapping), which is what VDST wants.
__device__ __forceinline__ void async_b128(const __bf16* gbase, unsigned goff_bytes, void* ldsdst) {
    unsigned l = (unsigned)(uintptr_t)ldsdst;
    asm volatile("global_load_async_to_lds_b128 %0, %1, %2"
                 :: "v"(l), "v"(goff_bytes), "s"(gbase)
                 : "memory");
}

// ---------------------------------------------------------------------------
// Kernel 1: fused QKV projection.
//   Wcat (640x512) @ x[b] (512x4096); rows 0..63 = Q, 64..127 = K, 128..639 = V
//   qws/kws: [B][N][64] bf16 row-major; vws: [B][C][N] bf16 channel-major.
// ---------------------------------------------------------------------------
__global__ __launch_bounds__(256)
void qkv_proj_kernel(const float* __restrict__ x,
                     const float* __restrict__ Wq, const float* __restrict__ bq,
                     const float* __restrict__ Wk, const float* __restrict__ bk,
                     const float* __restrict__ Wv, const float* __restrict__ bv,
                     __bf16* __restrict__ qws, __bf16* __restrict__ kws,
                     __bf16* __restrict__ vws)
{
    __shared__ alignas(16) __bf16 Alds[128 * 32];   // [r][c], c-contiguous
    __shared__ alignas(16) __bf16 Blds[128 * 32];   // [n][c], c-contiguous

    const int nt = blockIdx.x;          // 0..31 (n tile of 128)
    const int rt = blockIdx.y;          // 0..4  (r tile of 128; 5*128 = 640)
    const int b  = blockIdx.z;
    const int t    = threadIdx.x;
    const int w    = t >> 5;
    const int lane = t & 31;
    const int wu   = __builtin_amdgcn_readfirstlane(w);   // wave-uniform wave id
    const int r0 = rt * 128, n0 = nt * 128;
    const int hi = lane >> 4, col = lane & 15;

    v8f acc[8];
    const v8f vzero = {};
    for (int i = 0; i < 8; ++i) acc[i] = vzero;

    for (int kc = 0; kc < C_; kc += 32) {
        // Stage A: row index rr = wu + 8j is wave-uniform -> scalar branches
        // for the Wq/Wk/Wv split instead of per-lane selects.
        for (int j = 0; j < 16; ++j) {
            const int rr = wu + 8 * j;
            const int r  = r0 + rr;
            const int c  = kc + lane;
            float wv;
            if (r < 64)        wv = Wq[r * C_ + c];
            else if (r < 128)  wv = Wk[(r - 64) * C_ + c];
            else               wv = Wv[(r - 128) * C_ + c];
            Alds[rr * 32 + lane] = (__bf16)wv;
        }
        // Stage B transposed: x[b][kc..kc+31][n0..n0+127] -> Blds[n][c].
        for (int e = t; e < 128 * 32; e += 256) {
            const int cc = e >> 7, nn = e & 127;
            Blds[nn * 32 + cc] =
                (__bf16)x[(size_t)b * C_ * N_ + (size_t)(kc + cc) * N_ + (n0 + nn)];
        }
        __syncthreads();

        const v16bf afrag = load_frag16(Alds, w * 16 + col, 32, lane);
        for (int nt8 = 0; nt8 < 8; ++nt8) {
            const v16bf bfrag = load_frag16(Blds + nt8 * 16 * 32, col, 32, lane);
            acc[nt8] = __builtin_amdgcn_wmma_f32_16x16x32_bf16(
                false, afrag, false, bfrag, (short)0, acc[nt8], false, false);
        }
        __syncthreads();
    }

    // Epilogue: D layout: VGPR v -> row v + hi*8, col = lane&15.
    for (int nt8 = 0; nt8 < 8; ++nt8) {
        const int n = n0 + nt8 * 16 + col;
        for (int v = 0; v < 8; ++v) {
            const int r = r0 + w * 16 + v + hi * 8;
            const float bias = (r < 64) ? bq[r] : (r < 128 ? bk[r - 64] : bv[r - 128]);
            const __bf16 h = (__bf16)(acc[nt8][v] + bias);
            if (r < 64)
                qws[(size_t)b * N_ * DQK_ + (size_t)n * DQK_ + r] = h;
            else if (r < 128)
                kws[(size_t)b * N_ * DQK_ + (size_t)n * DQK_ + (r - 64)] = h;
            else
                vws[(size_t)b * C_ * N_ + (size_t)(r - 128) * N_ + n] = h;
        }
    }
}

// ---------------------------------------------------------------------------
// Kernel 2: flash attention + residual epilogue.
// Grid: (N/64 query tiles, B). Block: 512 threads = 16 waves.
//   wave w: qm = w&3 -> 16 query rows; cs = w>>2 -> 128-channel slab.
// Double-buffered async K/V staging: prefetch tile i+1 while computing tile i.
// ---------------------------------------------------------------------------
__global__ __launch_bounds__(512)
void flash_attn_kernel(const __bf16* __restrict__ qws, const __bf16* __restrict__ kws,
                       const __bf16* __restrict__ vws, const float* __restrict__ x,
                       const float* __restrict__ gamma, float* __restrict__ out)
{
    __shared__ alignas(16) __bf16 Klds[2][64 * 64];       // 2 x  8 KB [key][d]
    __shared__ alignas(16) __bf16 Vlds[2][512 * 64];      // 2 x 64 KB [c][key]
    __shared__ alignas(16) __bf16 Plds[16 * 16 * 64];     // 32 KB per-wave [16q][64key]

    const int qt = blockIdx.x;
    const int b  = blockIdx.y;
    const int t    = threadIdx.x;
    const int w    = t >> 5;
    const int lane = t & 31;
    const int qm = w & 3;
    const int cs = w >> 2;
    const int hi = lane >> 4, col = lane & 15;

    const __bf16* kbase_g = kws + (size_t)b * N_ * DQK_;
    const __bf16* vbase_g = vws + (size_t)b * C_ * N_;

    // Per-thread async staging of one 64-key tile: 1 (K) + 8 (V) b128 copies.
    const int ke   = t * 8;                 // K tile element base (64x64)
    const int kkey = ke >> 6, kd = ke & 63;

    // Q fragments for this wave's 16 rows (d = 64 -> two 32-wide k chunks).
    const __bf16* qbase = qws + (size_t)b * N_ * DQK_ + (size_t)(qt * 64 + qm * 16) * DQK_;
    const v16bf qf0 = load_frag16(qbase + 0,  col, DQK_, lane);
    const v16bf qf1 = load_frag16(qbase + 32, col, DQK_, lane);

    v8f acc[8];
    const v8f vzero = {};
    for (int i = 0; i < 8; ++i) acc[i] = vzero;
    float m_i[8], l_i[8];
    for (int v = 0; v < 8; ++v) { m_i[v] = -1e30f; l_i[v] = 0.f; }

    // Prologue: stage tile 0 into buffer 0.
    {
        async_b128(kbase_g, (unsigned)(((0 + kkey) * DQK_ + kd) * 2), &Klds[0][ke]);
        for (int i = 0; i < 8; ++i) {
            const int e = (i * 512 + t) * 8;
            const int c = e >> 6, kk = e & 63;
            async_b128(vbase_g, (unsigned)((c * N_ + 0 + kk) * 2), &Vlds[0][e]);
        }
    }

    const int NT = N_ / 64;
    for (int it = 0; it < NT; ++it) {
        const int cur = it & 1;
        if (it + 1 < NT) {
            // Prefetch next tile into the other buffer, then drain this tile's
            // 9 async copies (ASYNCcnt decrements in issue order).
            const int kt0n = (it + 1) * 64;
            async_b128(kbase_g, (unsigned)(((kt0n + kkey) * DQK_ + kd) * 2), &Klds[cur ^ 1][ke]);
            for (int i = 0; i < 8; ++i) {
                const int e = (i * 512 + t) * 8;
                const int c = e >> 6, kk = e & 63;
                async_b128(vbase_g, (unsigned)((c * N_ + kt0n + kk) * 2), &Vlds[cur ^ 1][e]);
            }
            asm volatile("s_wait_asynccnt 0x9" ::: "memory");
        } else {
            asm volatile("s_wait_asynccnt 0x0" ::: "memory");
        }
        __syncthreads();

        const __bf16* Kb = Klds[cur];
        const __bf16* Vb = Vlds[cur];

        // S = Q(16x64) @ K^T(64 keys): 4 score tiles, 2 WMMA each.
        v8f s[4];
        for (int kt = 0; kt < 4; ++kt) {
            v8f z = vzero;
            const v16bf bf0 = load_frag16(Kb + (kt * 16) * 64 + 0,  col, 64, lane);
            const v16bf bf1 = load_frag16(Kb + (kt * 16) * 64 + 32, col, 64, lane);
            z = __builtin_amdgcn_wmma_f32_16x16x32_bf16(false, qf0, false, bf0, (short)0, z, false, false);
            z = __builtin_amdgcn_wmma_f32_16x16x32_bf16(false, qf1, false, bf1, (short)0, z, false, false);
            s[kt] = z;
        }

        // Online softmax per query row (row = v + hi*8, cols over 16 lanes).
        __bf16* pbase = Plds + w * (16 * 64);
        for (int v = 0; v < 8; ++v) {
            float mx = fmaxf(fmaxf(s[0][v], s[1][v]), fmaxf(s[2][v], s[3][v]));
            for (int off = 1; off < 16; off <<= 1)
                mx = fmaxf(mx, __shfl_xor(mx, off, 32));
            const float mnew  = fmaxf(m_i[v], mx);
            const float scale = __expf(m_i[v] - mnew);
            float rs = 0.f;
            const int qrow = v + hi * 8;
            for (int kt = 0; kt < 4; ++kt) {
                const float p = __expf(s[kt][v] - mnew);
                rs += p;
                pbase[qrow * 64 + kt * 16 + col] = (__bf16)p;
            }
            for (int off = 1; off < 16; off <<= 1)
                rs += __shfl_xor(rs, off, 32);
            l_i[v] = l_i[v] * scale + rs;
            m_i[v] = mnew;
            for (int ct = 0; ct < 8; ++ct) acc[ct][v] *= scale;
        }
        // Wave-local P store -> load ordering (per-wave private LDS region).
        asm volatile("s_wait_dscnt 0x0" ::: "memory");

        // acc(16q x 128c) += P(16q x 64keys) @ V^T(64keys x 128c)
        for (int kk = 0; kk < 2; ++kk) {
            const v16bf pf = load_frag16(pbase + kk * 32, col, 64, lane);
            for (int ct = 0; ct < 8; ++ct) {
                const __bf16* vb = Vb + (size_t)(cs * 128 + ct * 16) * 64 + kk * 32;
                const v16bf vf = load_frag16(vb, col, 64, lane);
                acc[ct] = __builtin_amdgcn_wmma_f32_16x16x32_bf16(
                    false, pf, false, vf, (short)0, acc[ct], false, false);
            }
        }
        __syncthreads();   // readers done before buf 'cur' is refilled at it+1
    }

    // Epilogue: out = gamma * acc/l + x   (out[b][c][n], n = query index)
    const float g = gamma[0];
    for (int ct = 0; ct < 8; ++ct) {
        const int c = cs * 128 + ct * 16 + col;
        for (int v = 0; v < 8; ++v) {
            const int n = qt * 64 + qm * 16 + v + hi * 8;
            const size_t idx = (size_t)b * C_ * N_ + (size_t)c * N_ + n;
            out[idx] = g * (acc[ct][v] / l_i[v]) + x[idx];
        }
    }
}

// ---------------------------------------------------------------------------
extern "C" void kernel_launch(void* const* d_in, const int* in_sizes, int n_in,
                              void* d_out, int out_size, void* d_ws, size_t ws_size,
                              hipStream_t stream) {
    const float* x     = (const float*)d_in[0];
    const float* Wq    = (const float*)d_in[1];
    const float* bq    = (const float*)d_in[2];
    const float* Wk    = (const float*)d_in[3];
    const float* bk    = (const float*)d_in[4];
    const float* Wv    = (const float*)d_in[5];
    const float* bv    = (const float*)d_in[6];
    const float* gamma = (const float*)d_in[7];
    float* out = (float*)d_out;

    // Workspace: bf16 Q [B,N,64] + K [B,N,64] + V [B,C,N]  (~42 MB)
    __bf16* qws = (__bf16*)d_ws;
    __bf16* kws = qws + (size_t)B_ * N_ * DQK_;
    __bf16* vws = kws + (size_t)B_ * N_ * DQK_;

    dim3 g1(N_ / 128, 640 / 128, B_);
    qkv_proj_kernel<<<g1, 256, 0, stream>>>(x, Wq, bq, Wk, bk, Wv, bv, qws, kws, vws);

    dim3 g2(N_ / 64, B_);
    flash_attn_kernel<<<g2, 512, 0, stream>>>(qws, kws, vws, x, gamma, out);
}
